// KernelAttention2D_56977036148956
// MI455X (gfx1250) — matrix operand order repaired and checked by reading, not compiled, not run on hardware
//
#include <hip/hip_runtime.h>

typedef __attribute__((ext_vector_type(16))) _Float16 v16h;
typedef __attribute__((ext_vector_type(8)))  float    v8f;

#define B_ 8
#define T_ 2048
#define D_ 1024
#define INV_2SIG2 (1.0f / 512.0f)   // 1/(2*sigma^2), sigma=16

static __device__ __forceinline__ int iabs_(int v) { return v < 0 ? -v : v; }

// ---------------------------------------------------------------------------
// Pass 1: fp32 row normalizer, stored as reciprocal:
//   rdenom[b,t] = 1 / sum_s w[t,s]
// grid (T_/256, B_), block 256
// ---------------------------------------------------------------------------
__global__ __launch_bounds__(256) void ka2d_denom(const int* __restrict__ posv,
                                                  const int* __restrict__ posh,
                                                  float* __restrict__ rdenom) {
    __shared__ int   sp[T_];
    __shared__ float ev[64];
    const int b   = blockIdx.y;
    const int tid = threadIdx.x;

    for (int i = tid; i < T_; i += 256)
        sp[i] = (posv[b * T_ + i] & 0xFF) | ((posh[b * T_ + i] & 0xFF) << 8);
    if (tid < 64) {
        float k = (float)tid;
        ev[tid] = __expf(-(k * k) * INV_2SIG2);
    }
    __syncthreads();

    const int t  = blockIdx.x * 256 + tid;
    const int p  = sp[t];
    const int pv = p & 0xFF, ph = p >> 8;
    float sum = 0.0f;
    for (int s = 0; s < T_; ++s) {
        int q  = sp[s];
        int dv = iabs_((q & 0xFF) - pv);
        int dh = iabs_((q >> 8)   - ph);
        sum += ev[dv] * ev[dh];
    }
    rdenom[b * T_ + t] = 1.0f / sum;
}

// ---------------------------------------------------------------------------
// Pass 2: out[b,t,d] = (sum_s w[t,s] * x[b,s,d]) * rdenom[b,t]  via WMMA f16
// Block tile: M=64 (t) x N=128 (d). 256 threads = 8 waves; wave w owns
// d-subtile w*16 and accumulates 4 stacked 16x16 WMMA tiles (m = 0..3),
// reusing one B fragment for 4 A fragments -> 4x x-traffic reduction.
// grid (D_/128, T_/64, B_), block 256
// ---------------------------------------------------------------------------
__global__ __launch_bounds__(256) void ka2d_attn(const float* __restrict__ x,
                                                 const int*   __restrict__ posv,
                                                 const int*   __restrict__ posh,
                                                 const float* __restrict__ rdenom,
                                                 float*       __restrict__ out) {
    __shared__ int      sp[T_];
    __shared__ float    ev[64];
    __shared__ __align__(16) unsigned sa[64 * 16];   // 64 rows x 32 f16 weights

    const int b    = blockIdx.z;
    const int t0   = blockIdx.y * 64;
    const int d0   = blockIdx.x * 128;
    const int tid  = threadIdx.x;
    const int lane = tid & 31;          // wave32
    const int wave = tid >> 5;

    for (int i = tid; i < T_; i += 256)
        sp[i] = (posv[b * T_ + i] & 0xFF) | ((posh[b * T_ + i] & 0xFF) << 8);
    if (tid < 64) {
        float k = (float)tid;
        ev[tid] = __expf(-(k * k) * INV_2SIG2);
    }
    __syncthreads();

    // Cooperative A-tile build: 64x32 f16 = 1024 dwords, 4 dwords/thread
    // thread -> row = tid/4, k = (tid%4)*8 .. +7  (one aligned b128 store)
    const int arow = tid >> 2;          // 0..63
    const int akb  = (tid & 3) * 8;     // 0,8,16,24
    const int tp   = sp[t0 + arow];
    const int tpv  = tp & 0xFF, tph = tp >> 8;

    // Fragment geometry (ISA 16-bit WMMA layouts, wave32)
    const int half = lane >> 4;         // 0: lanes 0-15, 1: lanes 16-31
    const int col  = lane & 15;         // N index
    const int d    = d0 + wave * 16 + col;
    const int klo  = half * 16;         // B: lanes 16-31 hold K=8..15, 24..31 (klo unused)

    v8f acc0 = {}, acc1 = {}, acc2 = {}, acc3 = {};
    const float* xb = x + (size_t)b * T_ * D_ + d;

    for (int s0 = 0; s0 < T_; s0 += 32) {
        __syncthreads();   // previous A fragments consumed
        {
            unsigned pk[4];
#pragma unroll
            for (int j = 0; j < 4; ++j) {
                int q0 = sp[s0 + akb + 2 * j];
                int q1 = sp[s0 + akb + 2 * j + 1];
                float w0 = ev[iabs_((q0 & 0xFF) - tpv)] * ev[iabs_((q0 >> 8) - tph)];
                float w1 = ev[iabs_((q1 & 0xFF) - tpv)] * ev[iabs_((q1 >> 8) - tph)];
                unsigned short h0 = __builtin_bit_cast(unsigned short, (_Float16)w0);
                unsigned short h1 = __builtin_bit_cast(unsigned short, (_Float16)w1);
                pk[j] = (unsigned)h0 | ((unsigned)h1 << 16);
            }
            *(uint4*)&sa[arow * 16 + (akb >> 1)] =
                make_uint4(pk[0], pk[1], pk[2], pk[3]);
        }
        __syncthreads();   // A tile visible

        // B fragment: column d, element j of v16h = K 8*half+j (j<8), 16+8*half+(j-8) (j>=8)
        union BF { _Float16 h[16]; v16h v; } bf;
        const float* xp = xb + (size_t)(s0 + half * 8) * D_;
#pragma unroll
        for (int j = 0; j < 16; ++j)
            bf.h[j] = (_Float16)xp[(size_t)(j + (j >> 3) * 8) * D_];

        // A fragments: lanes 0-15 get K 0..7 & 16..23, lanes 16-31 K 8..15 & 24..31
        const uint4* sa4 = (const uint4*)sa;
        union AF { uint4 q[2]; v16h v; } af0, af1, af2, af3;
        const int ab = (lane & 15) * 4 + half;   // uint4 units within 16-row band
        af0.q[0] = sa4[ab];            af0.q[1] = sa4[ab + 2];
        af1.q[0] = sa4[ab + 16 * 4];   af1.q[1] = sa4[ab + 16 * 4 + 2];
        af2.q[0] = sa4[ab + 32 * 4];   af2.q[1] = sa4[ab + 32 * 4 + 2];
        af3.q[0] = sa4[ab + 48 * 4];   af3.q[1] = sa4[ab + 48 * 4 + 2];

        acc0 = __builtin_amdgcn_wmma_f32_16x16x32_f16(
            false, af0.v, false, bf.v, (short)0, acc0, false, false);
        acc1 = __builtin_amdgcn_wmma_f32_16x16x32_f16(
            false, af1.v, false, bf.v, (short)0, acc1, false, false);
        acc2 = __builtin_amdgcn_wmma_f32_16x16x32_f16(
            false, af2.v, false, bf.v, (short)0, acc2, false, false);
        acc3 = __builtin_amdgcn_wmma_f32_16x16x32_f16(
            false, af3.v, false, bf.v, (short)0, acc3, false, false);
    }

    // Normalize (multiply by precomputed reciprocal) & store.
    // C layout: VGPR r -> row r (lanes 0-15) / r+8 (lanes 16-31)
    const float* dn = rdenom + b * T_ + t0;
    float* ob = out + ((size_t)b * T_ + t0) * D_ + d;
    {
        v8f a[4] = {acc0, acc1, acc2, acc3};
#pragma unroll
        for (int m = 0; m < 4; ++m) {
#pragma unroll
            for (int r = 0; r < 8; ++r) {
                const int trow = m * 16 + r + half * 8;
                ob[(size_t)trow * D_] = a[m][r] * dn[trow];
            }
        }
    }
}

// ---------------------------------------------------------------------------
extern "C" void kernel_launch(void* const* d_in, const int* in_sizes, int n_in,
                              void* d_out, int out_size, void* d_ws, size_t ws_size,
                              hipStream_t stream) {
    const float* x    = (const float*)d_in[0];
    const int*   posv = (const int*)d_in[1];
    const int*   posh = (const int*)d_in[2];
    float*       out  = (float*)d_out;
    float*       dnm  = (float*)d_ws;          // B_*T_ fp32 reciprocals (64 KB)

    dim3 g1(T_ / 256, B_);
    ka2d_denom<<<g1, 256, 0, stream>>>(posv, posh, dnm);

    dim3 g2(D_ / 128, T_ / 64, B_);
    ka2d_attn<<<g2, 256, 0, stream>>>(x, posv, posh, dnm, out);
}
